// Attention_4123168604637
// MI455X (gfx1250) — matrix operand order repaired and checked by reading, not compiled, not run on hardware
//
#include <hip/hip_runtime.h>
#include <hip/hip_bf16.h>

#define BB   8
#define NN   1024
#define DIMM 512
#define HH   8
#define DD   64
#define TOPK 512

typedef __attribute__((ext_vector_type(16))) __bf16 v16bf;
typedef __attribute__((ext_vector_type(8)))  float  v8f;

union BFrag { v16bf v; unsigned int u[8]; uint4 q[2]; };

__device__ __forceinline__ unsigned int f2bf(float f) {
  unsigned int x = __float_as_uint(f);
  return (x + 0x7FFFu + ((x >> 16) & 1u)) >> 16;   // RNE to bf16
}
__device__ __forceinline__ unsigned int packbf(float lo, float hi) {
  return f2bf(lo) | (f2bf(hi) << 16);
}
// fast pack (round-half-up): 2 VALU per half
__device__ __forceinline__ unsigned int pkfast(float lo, float hi) {
  unsigned int l = (__float_as_uint(lo) + 0x8000u) >> 16;
  unsigned int h = (__float_as_uint(hi) + 0x8000u) & 0xFFFF0000u;
  return h | l;
}

// ---- fragment loaders: operands staged contiguous-in-K -> two b128 loads ----
// A 16x32 bf16 (ISA 7.12.2): lane = kg*16+m; VGPR0-3 = K [kg*8, kg*8+8), VGPR4-7 = +16
__device__ __forceinline__ v16bf frag_a_bf16(const unsigned short* rowptr, int k0) {
  int kg = (threadIdx.x >> 4) & 1;
  BFrag f;
  f.q[0] = *(const uint4*)(rowptr + k0 + kg * 8);
  f.q[1] = *(const uint4*)(rowptr + k0 + 16 + kg * 8);
  return f.v;
}
// B 32x16 bf16: lane = kg*16+n; VGPR0-3 = K [kg*8, kg*8+8), VGPR4-7 = +16 (matches A)
__device__ __forceinline__ v16bf frag_b_bf16(const unsigned short* rowptr, int k0) {
  int kg = (threadIdx.x >> 4) & 1;
  BFrag f;
  f.q[0] = *(const uint4*)(rowptr + k0 + kg * 8);
  f.q[1] = *(const uint4*)(rowptr + k0 + 16 + kg * 8);
  return f.v;
}
// A fragment from f32 LDS row (probability tile): 4x ds_load_b128 + fast pack
__device__ __forceinline__ v16bf frag_a_f32(const float* rowptr, int k0) {
  int kg = (threadIdx.x >> 4) & 1;
  float4 p0 = *(const float4*)(rowptr + k0 + kg * 8);
  float4 p1 = *(const float4*)(rowptr + k0 + kg * 8 + 4);
  float4 p2 = *(const float4*)(rowptr + k0 + 16 + kg * 8);
  float4 p3 = *(const float4*)(rowptr + k0 + 16 + kg * 8 + 4);
  BFrag f;
  f.u[0] = pkfast(p0.x, p0.y); f.u[1] = pkfast(p0.z, p0.w);
  f.u[2] = pkfast(p1.x, p1.y); f.u[3] = pkfast(p1.z, p1.w);
  f.u[4] = pkfast(p2.x, p2.y); f.u[5] = pkfast(p2.z, p2.w);
  f.u[6] = pkfast(p3.x, p3.y); f.u[7] = pkfast(p3.z, p3.w);
  return f.v;
}

__device__ __forceinline__ v8f wmma_bf16(v16bf a, v16bf b, v8f c) {
  return __builtin_amdgcn_wmma_f32_16x16x32_bf16(false, a, false, b, (short)0, c, false, false);
}
#define VZERO {0.f,0.f,0.f,0.f,0.f,0.f,0.f,0.f}

// ============ Prep: Q f32 -> bf16 (vectorized) ============
__global__ __launch_bounds__(256) void cvt_bf16_kernel(
    const float* __restrict__ src, unsigned short* __restrict__ dst, int n4) {
  int i = blockIdx.x * blockDim.x + threadIdx.x;
  if (i < n4) {
    float4 v = ((const float4*)src)[i];
    uint2 u; u.x = packbf(v.x, v.y); u.y = packbf(v.z, v.w);
    ((uint2*)dst)[i] = u;
  }
}

// ============ Prep: f32 [rows][cols] -> bf16 transposed [cols][rows] (LDS tiled) ============
__global__ __launch_bounds__(256) void transpose_bf16_kernel(
    const float* __restrict__ src, unsigned short* __restrict__ dstT, int rows, int cols) {
  __shared__ unsigned short tile[32][33];
  int tx = threadIdx.x & 31, ty = threadIdx.x >> 5;
  int c0 = blockIdx.x * 32, r0 = blockIdx.y * 32;
#pragma unroll
  for (int i = ty; i < 32; i += 8)
    tile[i][tx] = (unsigned short)f2bf(src[(size_t)(r0 + i) * cols + c0 + tx]);
  __syncthreads();
#pragma unroll
  for (int i = ty; i < 32; i += 8)
    dstT[(size_t)(c0 + i) * rows + r0 + tx] = tile[tx][i];
}

// ============ Kernel 1: qkv GEMM (16x64 per wave), scatter q*0.125 / k / v^T ============
__global__ __launch_bounds__(256) void qkv_gemm_kernel(
    const unsigned short* __restrict__ Qbf, const unsigned short* __restrict__ WqkvT,
    const float* __restrict__ bqkv,
    unsigned short* __restrict__ qbf, unsigned short* __restrict__ kbf,
    unsigned short* __restrict__ vT) {
  int wave = threadIdx.x >> 5, lane = threadIdx.x & 31;
  int mi = lane & 15, mh = lane >> 4;
  int m0 = blockIdx.y * 16;
  int n0 = (blockIdx.x * 8 + wave) * 64;           // 64 output columns per wave
  const unsigned short* arow = Qbf + (size_t)(m0 + mi) * DIMM;
  const unsigned short* brow0 = WqkvT + (size_t)(n0 + mi) * DIMM;
  v8f c[4] = {VZERO, VZERO, VZERO, VZERO};
  for (int k0 = 0; k0 < DIMM; k0 += 32) {
    v16bf a = frag_a_bf16(arow, k0);
#pragma unroll
    for (int t = 0; t < 4; ++t)
      c[t] = wmma_bf16(a, frag_b_bf16(brow0 + (size_t)t * 16 * DIMM, k0), c[t]);
  }
#pragma unroll
  for (int t = 0; t < 4; ++t) {
    int colbase = n0 + t * 16;                     // wave-uniform -> uniform branch
    int which = colbase >> 9;                      // 0=q 1=k 2=v
    int h  = (colbase & 511) >> 6;
    int dd = (colbase & 63) + mi;
    float bias = bqkv[colbase + mi];
#pragma unroll
    for (int i = 0; i < 8; ++i) {
      int row = m0 + i + 8 * mh;
      int b_ = row >> 10, nq = row & (NN - 1);
      float val = c[t][i] + bias;
      size_t bh = (size_t)(b_ * HH + h);
      if (which == 0)      qbf[(bh * NN + nq) * DD + dd] = (unsigned short)f2bf(val * 0.125f);
      else if (which == 1) kbf[(bh * NN + nq) * DD + dd] = (unsigned short)f2bf(val);
      else                 vT[(bh * DD + dd) * NN + nq]  = (unsigned short)f2bf(val);
    }
  }
}

// ============ Kernel 2: scores -> softmax -> head-mean weight + column sums ============
__global__ __launch_bounds__(256) void attn_weight_kernel(
    const unsigned short* __restrict__ qbf, const unsigned short* __restrict__ kbf,
    float* __restrict__ weight, float* __restrict__ colsum) {
  __shared__ float S[16][NN + 4];
  int b  = blockIdx.x >> 6;
  int q0 = (blockIdx.x & 63) * 16;
  int wave = threadIdx.x >> 5, lane = threadIdx.x & 31;
  int mi = lane & 15, mh = lane >> 4;
  float acc[64];
#pragma unroll
  for (int i = 0; i < 64; ++i) acc[i] = 0.f;

  for (int h = 0; h < HH; ++h) {
    const unsigned short* qb = qbf + (size_t)(b * HH + h) * NN * DD;
    const unsigned short* kb = kbf + (size_t)(b * HH + h) * NN * DD;
    const unsigned short* arow = qb + (size_t)(q0 + mi) * DD;
    v16bf a0 = frag_a_bf16(arow, 0), a1 = frag_a_bf16(arow, 32);  // hoisted A frags
    for (int ct = wave; ct < 64; ct += 8) {
      int n0 = ct * 16;
      const unsigned short* brow = kb + (size_t)(n0 + mi) * DD;   // k row = column of k^T
      v8f c = VZERO;
      c = wmma_bf16(a0, frag_b_bf16(brow, 0),  c);
      c = wmma_bf16(a1, frag_b_bf16(brow, 32), c);
#pragma unroll
      for (int i = 0; i < 8; ++i) S[i + 8 * mh][n0 + mi] = c[i];
    }
    __syncthreads();
#pragma unroll
    for (int rr = 0; rr < 2; ++rr) {               // wave owns rows 2w, 2w+1
      int row = wave * 2 + rr;
      float mx = -1e30f;
      for (int j = lane; j < NN; j += 32) mx = fmaxf(mx, S[row][j]);
#pragma unroll
      for (int off = 16; off >= 1; off >>= 1) mx = fmaxf(mx, __shfl_xor(mx, off, 32));
      float sum = 0.f;
      for (int j = lane; j < NN; j += 32) { float e = __expf(S[row][j] - mx); S[row][j] = e; sum += e; }
#pragma unroll
      for (int off = 16; off >= 1; off >>= 1) sum += __shfl_xor(sum, off, 32);
      float inv = 0.125f / sum;                    // fold head mean
      for (int j = lane; j < NN; j += 32) S[row][j] *= inv;
    }
    __syncthreads();
    int c0 = threadIdx.x * 4;
#pragma unroll
    for (int r = 0; r < 16; ++r)
#pragma unroll
      for (int j = 0; j < 4; ++j) acc[r * 4 + j] += S[r][c0 + j];
    __syncthreads();
  }
  int c0 = threadIdx.x * 4;
  float cs[4] = {0.f, 0.f, 0.f, 0.f};
#pragma unroll
  for (int r = 0; r < 16; ++r) {
    float4 v4 = make_float4(acc[r * 4], acc[r * 4 + 1], acc[r * 4 + 2], acc[r * 4 + 3]);
    *(float4*)(weight + ((size_t)b * NN + q0 + r) * NN + c0) = v4;
#pragma unroll
    for (int j = 0; j < 4; ++j) cs[j] += acc[r * 4 + j];
  }
#pragma unroll
  for (int j = 0; j < 4; ++j) atomicAdd(&colsum[b * NN + c0 + j], cs[j]);
}

// ============ Kernel 3: per-batch bitonic top-512 (desc, ties -> lower idx) ============
__global__ __launch_bounds__(512) void topk_kernel(
    const float* __restrict__ colsum, int* __restrict__ inds_ws, int* __restrict__ inds_out) {
  __shared__ float sval[NN];
  __shared__ int   sidx[NN];
  int b = blockIdx.x, t = threadIdx.x;
  for (int i = t; i < NN; i += 512) { sval[i] = colsum[b * NN + i]; sidx[i] = i; }
  for (int k = 2; k <= NN; k <<= 1) {
    for (int j = k >> 1; j > 0; j >>= 1) {
      __syncthreads();
      int i   = ((t & ~(j - 1)) << 1) | (t & (j - 1));
      int ixj = i | j;
      float v0 = sval[i], v1 = sval[ixj];
      int   i0 = sidx[i], i1 = sidx[ixj];
      bool before10 = (v1 > v0) || (v1 == v0 && i1 < i0);
      bool doswap = ((i & k) == 0) ? before10 : !before10;
      if (doswap) { sval[i] = v1; sval[ixj] = v0; sidx[i] = i1; sidx[ixj] = i0; }
    }
  }
  __syncthreads();
  inds_ws[b * TOPK + t]  = sidx[t];
  inds_out[b * TOPK + t] = sidx[t];
}

// ============ Kernel 4: gathered-row attention, O = P @ V ============
__global__ __launch_bounds__(256) void attn_out_kernel(
    const unsigned short* __restrict__ qbf, const unsigned short* __restrict__ kbf,
    const unsigned short* __restrict__ vT, const int* __restrict__ inds,
    unsigned short* __restrict__ Obf) {
  __shared__ float S[16][NN + 4];
  __shared__ int   qidx[16];
  __shared__ float Cred[4][32][8];
  int bx = blockIdx.x;
  int b  = bx >> 8;
  int h  = (bx >> 5) & 7;
  int s0 = (bx & 31) * 16;
  int wave = threadIdx.x >> 5, lane = threadIdx.x & 31;
  int mi = lane & 15, mh = lane >> 4;
  if (threadIdx.x < 16) qidx[threadIdx.x] = inds[b * TOPK + s0 + threadIdx.x];
  __syncthreads();
  const unsigned short* qb = qbf + (size_t)(b * HH + h) * NN * DD;
  const unsigned short* kb = kbf + (size_t)(b * HH + h) * NN * DD;
  const unsigned short* vb = vT  + (size_t)(b * HH + h) * DD * NN;
  const unsigned short* arow = qb + (size_t)qidx[mi] * DD;
  v16bf a0 = frag_a_bf16(arow, 0), a1 = frag_a_bf16(arow, 32);
  for (int ct = wave; ct < 64; ct += 8) {
    int n0 = ct * 16;
    const unsigned short* brow = kb + (size_t)(n0 + mi) * DD;
    v8f c = VZERO;
    c = wmma_bf16(a0, frag_b_bf16(brow, 0),  c);
    c = wmma_bf16(a1, frag_b_bf16(brow, 32), c);
#pragma unroll
    for (int i = 0; i < 8; ++i) S[i + 8 * mh][n0 + mi] = c[i];
  }
  __syncthreads();
#pragma unroll
  for (int rr = 0; rr < 2; ++rr) {
    int row = wave * 2 + rr;
    float mx = -1e30f;
    for (int j = lane; j < NN; j += 32) mx = fmaxf(mx, S[row][j]);
#pragma unroll
    for (int off = 16; off >= 1; off >>= 1) mx = fmaxf(mx, __shfl_xor(mx, off, 32));
    float sum = 0.f;
    for (int j = lane; j < NN; j += 32) { float e = __expf(S[row][j] - mx); S[row][j] = e; sum += e; }
#pragma unroll
    for (int off = 16; off >= 1; off >>= 1) sum += __shfl_xor(sum, off, 32);
    float inv = 1.f / sum;
    for (int j = lane; j < NN; j += 32) S[row][j] *= inv;
  }
  __syncthreads();
  // P(16x1024) @ V(1024x64): wave -> d-tile = wave&3, K-half = wave>>2
  int n0 = (wave & 3) * 16;
  int kh = wave >> 2;
  const unsigned short* brow = vb + (size_t)(n0 + mi) * NN;   // vT row: pairs along keys
  const float* prow = &S[mi][0];
  v8f c = VZERO;
  for (int k0 = kh * 512; k0 < kh * 512 + 512; k0 += 32)
    c = wmma_bf16(frag_a_f32(prow, k0), frag_b_bf16(brow, k0), c);
  if (kh == 1) {
#pragma unroll
    for (int i = 0; i < 8; ++i) Cred[wave & 3][lane][i] = c[i];
  }
  __syncthreads();
  if (kh == 0) {
#pragma unroll
    for (int i = 0; i < 8; ++i) {
      float val = c[i] + Cred[wave][lane][i];
      int row = s0 + i + 8 * mh;
      Obf[((size_t)b * TOPK + row) * DIMM + h * DD + n0 + mi] = (unsigned short)f2bf(val);
    }
  }
}

// ============ Kernel 5: out = O @ W_o + b_o (16x64 per wave) ============
__global__ __launch_bounds__(256) void out_gemm_kernel(
    const unsigned short* __restrict__ Obf, const unsigned short* __restrict__ WoT,
    const float* __restrict__ bo, float* __restrict__ out) {
  int wave = threadIdx.x >> 5, lane = threadIdx.x & 31;
  int mi = lane & 15, mh = lane >> 4;
  int m0 = blockIdx.x * 16;
  int n0 = wave * 64;                              // 8 waves cover all 512 columns
  const unsigned short* arow = Obf + (size_t)(m0 + mi) * DIMM;
  const unsigned short* brow0 = WoT + (size_t)(n0 + mi) * DIMM;
  v8f c[4] = {VZERO, VZERO, VZERO, VZERO};
  for (int k0 = 0; k0 < DIMM; k0 += 32) {
    v16bf a = frag_a_bf16(arow, k0);
#pragma unroll
    for (int t = 0; t < 4; ++t)
      c[t] = wmma_bf16(a, frag_b_bf16(brow0 + (size_t)t * 16 * DIMM, k0), c[t]);
  }
#pragma unroll
  for (int t = 0; t < 4; ++t) {
    float bias = bo[n0 + t * 16 + mi];
#pragma unroll
    for (int i = 0; i < 8; ++i) {
      int row = m0 + i + 8 * mh;
      out[(size_t)row * DIMM + n0 + t * 16 + mi] = c[t][i] + bias;
    }
  }
}

extern "C" void kernel_launch(void* const* d_in, const int* in_sizes, int n_in,
                              void* d_out, int out_size, void* d_ws, size_t ws_size,
                              hipStream_t stream) {
  (void)in_sizes; (void)n_in; (void)out_size; (void)ws_size;
  const float* Q    = (const float*)d_in[0];
  const float* Wqkv = (const float*)d_in[1];
  const float* bqkv = (const float*)d_in[2];
  const float* Wo   = (const float*)d_in[3];
  const float* bo   = (const float*)d_in[4];

  float* out_f  = (float*)d_out;                                 // [B,512,512]
  float* weight = out_f + (size_t)BB * TOPK * DIMM;              // [B,1024,1024]
  int*   inds_o = (int*)(weight + (size_t)BB * NN * NN);         // [B,512] int32

  char* ws = (char*)d_ws;
  size_t off = 0;
  unsigned short* qbf   = (unsigned short*)(ws + off); off += (size_t)BB * HH * NN * DD * 2;
  unsigned short* kbf   = (unsigned short*)(ws + off); off += (size_t)BB * HH * NN * DD * 2;
  unsigned short* vT    = (unsigned short*)(ws + off); off += (size_t)BB * HH * NN * DD * 2;
  unsigned short* Obf   = (unsigned short*)(ws + off); off += (size_t)BB * TOPK * DIMM * 2;
  unsigned short* Qbf   = (unsigned short*)(ws + off); off += (size_t)BB * NN * DIMM * 2;
  unsigned short* WqkvT = (unsigned short*)(ws + off); off += (size_t)3 * DIMM * DIMM * 2;
  unsigned short* WoT   = (unsigned short*)(ws + off); off += (size_t)DIMM * DIMM * 2;
  float*          csum  = (float*)(ws + off);          off += (size_t)BB * NN * 4;
  int*            inds  = (int*)(ws + off);            off += (size_t)BB * TOPK * 4;

  hipMemsetAsync(csum, 0, (size_t)BB * NN * 4, stream);

  int nq4 = (BB * NN * DIMM) / 4;
  cvt_bf16_kernel<<<dim3((nq4 + 255) / 256), 256, 0, stream>>>(Q, Qbf, nq4);
  transpose_bf16_kernel<<<dim3(3 * DIMM / 32, DIMM / 32), 256, 0, stream>>>(Wqkv, WqkvT, DIMM, 3 * DIMM);
  transpose_bf16_kernel<<<dim3(DIMM / 32, DIMM / 32), 256, 0, stream>>>(Wo, WoT, DIMM, DIMM);

  qkv_gemm_kernel<<<dim3(3, 512), 256, 0, stream>>>(Qbf, WqkvT, bqkv, qbf, kbf, vT);
  attn_weight_kernel<<<dim3(BB * 64), 256, 0, stream>>>(qbf, kbf, weight, csum);
  topk_kernel<<<dim3(BB), 512, 0, stream>>>(csum, inds, inds_o);
  attn_out_kernel<<<dim3(BB * HH * 32), 256, 0, stream>>>(qbf, kbf, vT, inds, Obf);
  out_gemm_kernel<<<dim3(BB * TOPK / 16), 256, 0, stream>>>(Obf, WoT, bo, out_f);
}
